// SelfAttention_47090021433664
// MI455X (gfx1250) — matrix operand order repaired and checked by reading, not compiled, not run on hardware
//
#include <hip/hip_runtime.h>
#include <stdint.h>

#define B_ 4
#define C_ 256
#define N_ 4096

typedef __attribute__((ext_vector_type(16))) __bf16 v16bf;
typedef __attribute__((ext_vector_type(8)))  float  v8f;
typedef __attribute__((ext_vector_type(4)))  int    v4i;

union V16 { v16bf v; uint4 q[2]; };

#if defined(__AMDGCN__) && __has_builtin(__builtin_amdgcn_global_load_async_to_lds_b128) && __has_builtin(__builtin_amdgcn_s_wait_asynccnt)
#define HAVE_ASYNC 1
#else
#define HAVE_ASYNC 0
#endif

#if HAVE_ASYNC
#define GAS __attribute__((address_space(1)))
#define LAS __attribute__((address_space(3)))
#endif

// copy 16 bytes global -> LDS (async DMA when available)
__device__ __forceinline__ void cp16(const __bf16* g, __bf16* l) {
#if HAVE_ASYNC
  __builtin_amdgcn_global_load_async_to_lds_b128((GAS v4i*)g, (LAS v4i*)l, 0, 0);
#else
  *(uint4*)l = *(const uint4*)g;
#endif
}

__device__ __forceinline__ void wait_async() {
#if HAVE_ASYNC
  __builtin_amdgcn_s_wait_asynccnt(0);
#endif
}

__device__ __forceinline__ __bf16 f2bf(float f) {
  union { float f; uint32_t u; } c; c.f = f;
  uint32_t u = c.u;
  uint32_t r = (u + 0x7FFFu + ((u >> 16) & 1u)) >> 16;  // round-to-nearest-even
  union { uint16_t s; __bf16 b; } o; o.s = (uint16_t)r;
  return o.b;
}

// ---------------- Kernel 0: f32 -> bf16 convert (weights) ----------------
__global__ void cvt_f32_bf16(const float* __restrict__ in, __bf16* __restrict__ out, int n) {
  int i = blockIdx.x * 256 + threadIdx.x;
  if (i < n) out[i] = f2bf(in[i]);
}

// ---------------- Kernel 1: x [B][C][N] f32 -> xT [B][N][C] bf16 ----------------
__global__ __launch_bounds__(256) void xpose_f32_to_bf16(const float* __restrict__ x,
                                                         __bf16* __restrict__ xT) {
  __shared__ float tile[32][33];
  int b  = blockIdx.z;
  int n0 = blockIdx.x * 32;
  int c0 = blockIdx.y * 32;
  int tx = threadIdx.x, ty = threadIdx.y;  // 32 x 8
  const float* xp = x + (size_t)b * C_ * N_;
  #pragma unroll
  for (int k = 0; k < 32; k += 8)
    tile[ty + k][tx] = xp[(size_t)(c0 + ty + k) * N_ + n0 + tx];
  __syncthreads();
  __bf16* op = xT + (size_t)b * N_ * C_;
  #pragma unroll
  for (int k = 0; k < 32; k += 8)
    op[(size_t)(n0 + ty + k) * C_ + c0 + tx] = f2bf(tile[tx][ty + k]);
}

// ---------------- Kernel 3: bf16 transpose in[b][R][Cc] -> out[b][Cc][R] ----------------
__global__ __launch_bounds__(256) void xpose_bf16(const __bf16* __restrict__ in,
                                                  __bf16* __restrict__ out, int R, int Cc) {
  __shared__ __bf16 tile[32][33];
  int b  = blockIdx.z;
  int c0 = blockIdx.x * 32;
  int r0 = blockIdx.y * 32;
  int tx = threadIdx.x, ty = threadIdx.y;
  const __bf16* ip = in + (size_t)b * R * Cc;
  #pragma unroll
  for (int k = 0; k < 32; k += 8)
    tile[ty + k][tx] = ip[(size_t)(r0 + ty + k) * Cc + c0 + tx];
  __syncthreads();
  __bf16* op = out + (size_t)b * Cc * R;
  #pragma unroll
  for (int k = 0; k < 32; k += 8)
    op[(size_t)(c0 + ty + k) * R + r0 + tx] = tile[tx][ty + k];
}

// ---------------- Kernel 2: QKV projection via WMMA ----------------
__global__ __launch_bounds__(256) void qkv_proj(
    const __bf16* __restrict__ xT,
    const __bf16* __restrict__ Wq, const __bf16* __restrict__ Wk, const __bf16* __restrict__ Wv,
    const float* __restrict__ bq, const float* __restrict__ bk, const float* __restrict__ bv,
    __bf16* __restrict__ q, __bf16* __restrict__ k, __bf16* __restrict__ v) {
  int b    = blockIdx.y;
  int wave = threadIdx.x >> 5;
  int lane = threadIdx.x & 31;
  int lh = lane >> 4, lm = lane & 15;
  int nt = blockIdx.x * 8 + wave;
  int i  = nt * 16 + lm;

  // A-matrix (16x32 bf16): lane<16 holds K{0..7,16..23}, lane>=16 K{8..15,24..31}
  const __bf16* xrow = xT + (size_t)(b * N_ + i) * C_;
  V16 A[8];
  #pragma unroll
  for (int kc = 0; kc < 8; ++kc) {
    int c0 = kc * 32 + lh * 8;
    A[kc].q[0] = *(const uint4*)(xrow + c0);
    A[kc].q[1] = *(const uint4*)(xrow + c0 + 16);
  }

  const __bf16* Ws[3]  = {Wq, Wk, Wv};
  const float*  bs[3]  = {bq, bk, bv};
  __bf16*       os_[3] = {q, k, v};
  #pragma unroll
  for (int s = 0; s < 3; ++s) {
    const __bf16* W = Ws[s];
    __bf16* op = os_[s] + (size_t)b * N_ * C_;
    #pragma unroll 1
    for (int ot = 0; ot < 16; ++ot) {
      int o = ot * 16 + lm;
      float bval = bs[s][o];
      v8f acc = {};
      #pragma unroll
      for (int kc = 0; kc < 8; ++kc) {
        // B-matrix (32x16): lane<16 col o K{0..7,16..23} ; lane>=16 col o K{8..15,24..31}
        const __bf16* wrow = W + (size_t)o * C_ + kc * 32 + lh * 8;
        V16 Bv;
        Bv.q[0] = *(const uint4*)(wrow);
        Bv.q[1] = *(const uint4*)(wrow + 16);
        acc = __builtin_amdgcn_wmma_f32_16x16x32_bf16(false, A[kc].v, false, Bv.v,
                                                      (short)0, acc, false, false);
      }
      #pragma unroll
      for (int r = 0; r < 8; ++r) {
        int m = r + 8 * lh;
        op[(size_t)(nt * 16 + m) * C_ + o] = f2bf(acc[r] + bval);
      }
    }
  }
}

// ---------------- Kernel 4: flash attention + fused gamma*out + x ----------------
// wave -> (16-query tile, 128-channel half). Block = 8 waves shares double-buffered K/V tiles.
__global__ __launch_bounds__(256) void attn(
    const __bf16* __restrict__ qT, const __bf16* __restrict__ kT, const __bf16* __restrict__ vT,
    const float* __restrict__ x, const float* __restrict__ gamma, float* __restrict__ out) {
  // layout: [0,32K) = K double buffer, [32K,64K) = V double buffer, [64K,72K) = P staging
  __shared__ __align__(16) unsigned char smem[2 * 16384 + 2 * 16384 + 8192];  // 72KB

  const int b    = blockIdx.y;
  const int t    = threadIdx.x;
  const int wave = t >> 5;
  const int lane = t & 31;
  const int lh = lane >> 4, lm = lane & 15;
  const int qt   = blockIdx.x * 4 + (wave >> 1);
  const int half = wave & 1;
  const int ch0  = half * 128;

  const __bf16* kbase = kT + (size_t)b * N_ * C_;
  const __bf16* vbase = vT + (size_t)b * C_ * N_;
  __bf16* sPw = (__bf16*)(smem + 65536) + wave * 512;  // per-wave 16x32 bf16

  // Q tile in registers (A-layout, 8 K-chunks of 32)
  const __bf16* qrow = qT + (size_t)(b * N_ + qt * 16 + lm) * C_;
  V16 QA[8];
  #pragma unroll
  for (int kc = 0; kc < 8; ++kc) {
    int c0 = kc * 32 + lh * 8;
    QA[kc].q[0] = *(const uint4*)(qrow + c0);
    QA[kc].q[1] = *(const uint4*)(qrow + c0 + 16);
  }

  float m_r[8], l_r[8];
  v8f O[8];
  #pragma unroll
  for (int r = 0; r < 8; ++r) { m_r[r] = -1e30f; l_r[r] = 0.0f; }
  #pragma unroll
  for (int ct = 0; ct < 8; ++ct) { v8f z = {}; O[ct] = z; }

  const float LOG2E = 1.4426950408889634f;
  const int NT = N_ / 32;

  // stage tile jt into buffer buf: K 16KB (contiguous) + V 16KB (256 rows x 64B)
  auto stage = [&](int jt, int buf) {
    const __bf16* ks = kbase + (size_t)jt * 32 * C_;
    __bf16* kd = (__bf16*)(smem + (size_t)buf * 16384);
    #pragma unroll
    for (int u = 0; u < 4; ++u)
      cp16(ks + (size_t)(t + u * 256) * 8, kd + (size_t)(t + u * 256) * 8);
    const __bf16* vs = vbase + (size_t)t * N_ + jt * 32;
    __bf16* vd = (__bf16*)(smem + 32768 + (size_t)buf * 16384) + t * 32;
    #pragma unroll
    for (int u = 0; u < 4; ++u) cp16(vs + u * 8, vd + u * 8);
  };

  stage(0, 0);

  #pragma unroll 1
  for (int jt = 0; jt < NT; ++jt) {
    const int buf = jt & 1;
    wait_async();          // this wave's staged data has landed in LDS
    __syncthreads();       // everyone's data landed; prev buffer's readers done
    if (jt + 1 < NT) {
#if !HAVE_ASYNC
      __builtin_prefetch(kbase + (size_t)(jt + 1) * 32 * C_ + t * 64, 0, 1);
#endif
      stage(jt + 1, buf ^ 1);
    }
    const __bf16* sK = (const __bf16*)(smem + (size_t)buf * 16384);
    const __bf16* sV = (const __bf16*)(smem + 32768 + (size_t)buf * 16384);

    // S = Q . K^T : two 16x16 f32 tiles (j 0-15, 16-31)
    v8f S0 = {}, S1 = {};
    #pragma unroll
    for (int kc = 0; kc < 8; ++kc) {
      const __bf16* p0 = sK + (size_t)lm * C_        + kc * 32 + lh * 8;
      const __bf16* p1 = sK + (size_t)(16 + lm) * C_ + kc * 32 + lh * 8;
      V16 B0, B1;
      B0.q[0] = *(const uint4*)p0; B0.q[1] = *(const uint4*)(p0 + 16);
      B1.q[0] = *(const uint4*)p1; B1.q[1] = *(const uint4*)(p1 + 16);
      S0 = __builtin_amdgcn_wmma_f32_16x16x32_bf16(false, QA[kc].v, false, B0.v, (short)0, S0, false, false);
      S1 = __builtin_amdgcn_wmma_f32_16x16x32_bf16(false, QA[kc].v, false, B1.v, (short)0, S1, false, false);
    }

    // online softmax: row (i = r + 8*lh) stats, reduce across 16 lanes (j)
    float pscale[8];
    #pragma unroll
    for (int r = 0; r < 8; ++r) {
      float mx = fmaxf(S0[r], S1[r]);
      mx = fmaxf(mx, __shfl_xor(mx, 1, 32));
      mx = fmaxf(mx, __shfl_xor(mx, 2, 32));
      mx = fmaxf(mx, __shfl_xor(mx, 4, 32));
      mx = fmaxf(mx, __shfl_xor(mx, 8, 32));
      float mn = fmaxf(m_r[r], mx);
      float sc = exp2f((m_r[r] - mn) * LOG2E);
      m_r[r] = mn;
      pscale[r] = sc;
      float e0 = exp2f((S0[r] - mn) * LOG2E);
      float e1 = exp2f((S1[r] - mn) * LOG2E);
      S0[r] = e0; S1[r] = e1;
      float rs = e0 + e1;
      rs += __shfl_xor(rs, 1, 32);
      rs += __shfl_xor(rs, 2, 32);
      rs += __shfl_xor(rs, 4, 32);
      rs += __shfl_xor(rs, 8, 32);
      l_r[r] = l_r[r] * sc + rs;
    }
    #pragma unroll
    for (int ct = 0; ct < 8; ++ct) {
      #pragma unroll
      for (int r = 0; r < 8; ++r) O[ct][r] *= pscale[r];
    }

    // P (D-layout) -> per-wave LDS [i][j] -> reload in A-layout
    #pragma unroll
    for (int r = 0; r < 8; ++r) {
      int i = r + 8 * lh;
      sPw[i * 32 + lm]      = f2bf(S0[r]);
      sPw[i * 32 + 16 + lm] = f2bf(S1[r]);
    }
    __syncthreads();
    V16 PA;
    const __bf16* pp = sPw + lm * 32 + lh * 8;
    PA.q[0] = *(const uint4*)pp;
    PA.q[1] = *(const uint4*)(pp + 16);

    // O += P . V^T for this wave's 128 channels
    #pragma unroll
    for (int ct = 0; ct < 8; ++ct) {
      const __bf16* vp = sV + (size_t)(ch0 + ct * 16 + lm) * 32 + lh * 8;
      V16 Bv;
      Bv.q[0] = *(const uint4*)vp; Bv.q[1] = *(const uint4*)(vp + 16);
      O[ct] = __builtin_amdgcn_wmma_f32_16x16x32_bf16(false, PA.v, false, Bv.v, (short)0, O[ct], false, false);
    }
  }

  // epilogue: O /= l ; out[b][c][i] = gamma*O + x[b][c][i]
  float inv_l[8];
  #pragma unroll
  for (int r = 0; r < 8; ++r) inv_l[r] = 1.0f / l_r[r];
  float g = gamma[0];
  float* fb = (float*)sPw;  // 16x16 f32 bounce buffer
  const float* xb = x + (size_t)b * C_ * N_;
  float* ob = out + (size_t)b * C_ * N_;
  #pragma unroll 1
  for (int ct = 0; ct < 8; ++ct) {
    __syncthreads();
    #pragma unroll
    for (int r = 0; r < 8; ++r) {
      int m = r + 8 * lh;
      fb[lm * 16 + m] = O[ct][r] * inv_l[r];  // [c_local][i_local]
    }
    __syncthreads();
    int cl = lane >> 1;
    int il = (lane & 1) * 8;
    const float* row = fb + cl * 16 + il;
    float4 o0 = *(const float4*)row;
    float4 o1 = *(const float4*)(row + 4);
    size_t ga = (size_t)(ch0 + ct * 16 + cl) * N_ + qt * 16 + il;
    float4 x0 = *(const float4*)(xb + ga);
    float4 x1 = *(const float4*)(xb + ga + 4);
    float4 r0, r1;
    r0.x = g * o0.x + x0.x; r0.y = g * o0.y + x0.y; r0.z = g * o0.z + x0.z; r0.w = g * o0.w + x0.w;
    r1.x = g * o1.x + x1.x; r1.y = g * o1.y + x1.y; r1.z = g * o1.z + x1.z; r1.w = g * o1.w + x1.w;
    *(float4*)(ob + ga)     = r0;
    *(float4*)(ob + ga + 4) = r1;
  }
}

// ---------------- host launch ----------------
extern "C" void kernel_launch(void* const* d_in, const int* in_sizes, int n_in,
                              void* d_out, int out_size, void* d_ws, size_t ws_size,
                              hipStream_t stream) {
  const float* x     = (const float*)d_in[0];
  const float* Wq    = (const float*)d_in[1];
  const float* bq    = (const float*)d_in[2];
  const float* Wk    = (const float*)d_in[3];
  const float* bk    = (const float*)d_in[4];
  const float* Wv    = (const float*)d_in[5];
  const float* bv    = (const float*)d_in[6];
  const float* gamma = (const float*)d_in[7];
  float* out = (float*)d_out;

  const size_t SZ = (size_t)B_ * N_ * C_ * sizeof(__bf16);  // 8 MB per tensor
  char* ws = (char*)d_ws;
  __bf16* xT   = (__bf16*)(ws + 0 * SZ);
  __bf16* qT   = (__bf16*)(ws + 1 * SZ);
  __bf16* kT   = (__bf16*)(ws + 2 * SZ);
  __bf16* vtmp = (__bf16*)(ws + 3 * SZ);
  __bf16* vT   = (__bf16*)(ws + 4 * SZ);
  __bf16* Wqb  = (__bf16*)(ws + 5 * SZ);
  __bf16* Wkb  = (__bf16*)(ws + 5 * SZ + 131072);
  __bf16* Wvb  = (__bf16*)(ws + 5 * SZ + 262144);

  dim3 b256(256);
  const int nw = C_ * C_;
  cvt_f32_bf16<<<(nw + 255) / 256, b256, 0, stream>>>(Wq, Wqb, nw);
  cvt_f32_bf16<<<(nw + 255) / 256, b256, 0, stream>>>(Wk, Wkb, nw);
  cvt_f32_bf16<<<(nw + 255) / 256, b256, 0, stream>>>(Wv, Wvb, nw);

  dim3 tb(32, 8);
  xpose_f32_to_bf16<<<dim3(N_ / 32, C_ / 32, B_), tb, 0, stream>>>(x, xT);

  qkv_proj<<<dim3(N_ / 128, B_), b256, 0, stream>>>(xT, Wqb, Wkb, Wvb, bq, bk, bv, qT, kT, vtmp);

  // v [B][N][C] -> vT [B][C][N]
  xpose_bf16<<<dim3(C_ / 32, N_ / 32, B_), tb, 0, stream>>>(vtmp, vT, N_, C_);

  attn<<<dim3(N_ / 64, B_), b256, 0, stream>>>(qT, kT, vT, x, gamma, out);
}
